// ONNBlock_43095701848082
// MI455X (gfx1250) — matrix operand order repaired and checked by reading, not compiled, not run on hardware
//
#include <hip/hip_runtime.h>

#define BATCH  256
#define NBITS  64
#define TFEATS 2080          // 64*65/2, = 65 * 32
#define HDIM   4096
#define KSEEDS 640           // 2*N*(1+M)

typedef __attribute__((ext_vector_type(16))) _Float16 v16h;
typedef __attribute__((ext_vector_type(8)))  float    v8f;
typedef __attribute__((ext_vector_type(4)))  int      i32x4;

// ---- CDNA5 async global->LDS copy (ASYNCcnt-tracked), with safe fallback ----
#if defined(__has_builtin)
#  if __has_builtin(__builtin_amdgcn_global_load_async_to_lds_b128) && \
      __has_builtin(__builtin_amdgcn_s_wait_asynccnt)
#    define HAVE_ASYNC_LDS 1
#  endif
#endif
#ifndef HAVE_ASYNC_LDS
#  define HAVE_ASYNC_LDS 0
#endif

struct alignas(16) U4 { unsigned v[4]; };

__device__ __forceinline__ void cp16_g2l(const unsigned* g, unsigned* l) {
#if HAVE_ASYNC_LDS
    // Per-lane 16B async copy: emits global_load_async_to_lds_b128.
    __builtin_amdgcn_global_load_async_to_lds_b128(
        (__attribute__((address_space(1))) i32x4*)g,
        (__attribute__((address_space(3))) i32x4*)l,
        0, 0);
#else
    *(U4*)l = *(const U4*)g;
#endif
}

__device__ __forceinline__ void wait_async_copies() {
#if HAVE_ASYNC_LDS
    __builtin_amdgcn_s_wait_asynccnt(0);
#endif
}

__device__ __forceinline__ unsigned long long rotr64(unsigned long long x, int k) {
    k &= 63;
    return (x >> k) | (x << ((64 - k) & 63));
}

// ---------------------------------------------------------------------------
// Stage A: build 640 seed words per batch row, run XNOR triangle (word ops),
// sum bits per triangle position via wave32 ballot+popcount, emit f16 feats.
// One workgroup (8 waves) per batch row.
// ---------------------------------------------------------------------------
__global__ __launch_bounds__(256) void onn_feats_kernel(const int* __restrict__ x,
                                                        const int* __restrict__ masks,
                                                        _Float16* __restrict__ feats) {
    __shared__ unsigned long long s_x;
    __shared__ unsigned long long s_m[4];
    __shared__ int s_cnt[TFEATS];
    const int tid = threadIdx.x;
    const int b   = blockIdx.x;

    for (int t = tid; t < TFEATS; t += 256) s_cnt[t] = 0;
    if (tid == 0) {
        unsigned long long w = 0;
        for (int j = 0; j < NBITS; ++j)
            w |= ((unsigned long long)(x[b * NBITS + j] & 1)) << j;
        s_x = w;
        for (int m = 0; m < 4; ++m) {
            unsigned long long mw = 0;
            for (int j = 0; j < NBITS; ++j)
                mw |= ((unsigned long long)(masks[m * NBITS + j] & 1)) << j;
            s_m[m] = mw;
        }
    }
    __syncthreads();

    const unsigned long long xb = s_x;
    const int wave = tid >> 5;
    const int lane = tid & 31;

    // 640 seeds = 20 chunks of 32 (one seed per lane); mean over K is
    // order-invariant so (kb, mask) enumeration order is free.
    for (int c = wave; c < (KSEEDS / 32); c += 8) {
        const int s  = c * 32 + lane;
        const int kb = s & 127;       // base seed index
        const int mv = s >> 7;        // 0 = unmasked, 1..4 = mask (mv-1)
        unsigned long long seed =
            (kb < 64) ? rotr64(xb, kb) : __brevll(rotr64(xb, kb - 64));
        if (mv > 0) seed ^= s_m[mv - 1];

        unsigned long long cur = seed;
        int off = 0;
        for (int r = 0; r < NBITS; ++r) {
            const int L = NBITS - r;
            for (int j = 0; j < L; ++j) {
                unsigned bal = __builtin_amdgcn_ballot_w32(((cur >> j) & 1ull) != 0);
                if (lane == 0) atomicAdd(&s_cnt[off + j], (int)__popc(bal));
            }
            off += L;
            cur = ~(cur ^ (cur >> 1));                       // XNOR row update
            cur &= (L > 1) ? ((1ull << (L - 1)) - 1ull) : 0ull;
        }
    }
    __syncthreads();

    const float inv = 1.0f / (float)KSEEDS;
    for (int t = tid; t < TFEATS; t += 256) {
        float f = 2.0f * ((float)s_cnt[t] * inv) - 1.0f;
        feats[(size_t)b * TFEATS + t] = (_Float16)f;
    }
}

// ---------------------------------------------------------------------------
// Convert f32 weights (K x Nc, row-major) into chunk-swizzled packed f16:
// for K-chunk kc (32 K = 16 pairs) and column n, the 16 pair-dwords are
// contiguous:  Wfr[(kc*Nc + n)*16 + p] = { f16(W[32kc+2p][n]), f16(W[32kc+2p+1][n]) }
// so the GEMM's B tile fill and B fragment reads are fully contiguous.
// ---------------------------------------------------------------------------
__global__ __launch_bounds__(256) void onn_pack_w_kernel(const float* __restrict__ W,
                                                         unsigned* __restrict__ Wfr,
                                                         int Nc) {
    const int n  = blockIdx.x * 256 + threadIdx.x;
    const int kc = blockIdx.y;
    union { U4 q[4]; unsigned u[16]; } outv;
#pragma unroll
    for (int p = 0; p < 16; ++p) {
        union { _Float16 h[2]; unsigned u; } pk;
        pk.h[0] = (_Float16)W[(size_t)(kc * 32 + 2 * p)     * Nc + n];
        pk.h[1] = (_Float16)W[(size_t)(kc * 32 + 2 * p + 1) * Nc + n];
        outv.u[p] = pk.u;
    }
    U4* dst = (U4*)(Wfr + ((size_t)kc * Nc + n) * 16);
#pragma unroll
    for (int q = 0; q < 4; ++q) dst[q] = outv.q[q];
}

// ---------------------------------------------------------------------------
// f16 WMMA GEMM: C(M x Nn) = relu(A(M x K) * B(K x Nn) + bias), f32 accumulate.
// A: row-major f16 viewed as K-pairs (ldaP = K/2). B: chunk-swizzled (above).
// WG tile 32(M) x 256(N); 8 waves in 2x4, each wave 16x64 (4 accumulators).
// K stepped by 32; double-buffered LDS, async global->LDS b128 copies
// pipelined one K-chunk ahead (one s_wait_asynccnt + one barrier per step).
// All fragments live in distinct registers so the 10 ds_load_b128 per step
// can be issued ahead and waited on incrementally while WMMAs execute.
// ---------------------------------------------------------------------------
__global__ __launch_bounds__(256) void onn_gemm_kernel(const unsigned* __restrict__ Apk,
                                                       const unsigned* __restrict__ Bfr,
                                                       const float* __restrict__ bias,
                                                       _Float16* __restrict__ Cout,
                                                       int Nn, int K) {
    // Row strides padded to 20 dwords (80B): 16B-aligned and gcd(20,64)=4 puts
    // the 16 rows/cols touched by a half-wave on 16 distinct bank groups.
    __shared__ alignas(16) unsigned Al[2][32][20];    // [buf][row][pair]
    __shared__ alignas(16) unsigned Bl[2][256][20];   // [buf][col][pair]

    const int tid   = threadIdx.x;
    const int lane  = tid & 31;
    const int wv    = tid >> 5;
    const int waveM = wv >> 2;         // 0..1
    const int waveN = wv & 3;          // 0..3
    const int rowBase = blockIdx.y * 32;
    const int colBase = blockIdx.x * 256;
    const int ldaP = K >> 1;
    const int KC   = K >> 5;
    const int hi   = lane >> 4;        // half-wave select
    const int l15  = lane & 15;

    // Issue all b128 copies for one 32-wide K-chunk into LDS buffer `buf`.
    auto issue = [&](int buf, int kcn) {
        // B tile: each thread copies one column's 16 pair-dwords (64B, contig).
        const unsigned* srcB = Bfr + ((size_t)kcn * Nn + colBase + tid) * 16;
#pragma unroll
        for (int q = 0; q < 4; ++q)
            cp16_g2l(srcB + (q << 2), &Bl[buf][tid][q << 2]);
        // A tile: 32 rows x 16 pairs = 128 x b128, threads 0..127.
        if (tid < 128) {
            const int row = tid >> 2;
            const int q   = tid & 3;
            cp16_g2l(Apk + (size_t)(rowBase + row) * ldaP + ((kcn << 4) + (q << 2)),
                     &Al[buf][row][q << 2]);
        }
    };

    union Frag { U4 q[2]; unsigned u[8]; v16h v; };

    const v8f vzero = {0.f, 0.f, 0.f, 0.f, 0.f, 0.f, 0.f, 0.f};
    v8f acc0 = vzero, acc1 = vzero, acc2 = vzero, acc3 = vzero;

    issue(0, 0);   // prologue: chunk 0 into buffer 0

    for (int kc = 0; kc < KC; ++kc) {
        const int buf = kc & 1;
        wait_async_copies();     // this wave's copies into `buf` are done
        __syncthreads();         // ... and every other wave's too
        if (kc + 1 < KC) issue(buf ^ 1, kc + 1);   // overlap next chunk

        // A fragment (16-bit A 16x32 layout): lanes 0-15 -> pairs {0-3, 8-11},
        // lanes 16-31 -> pairs {4-7, 12-15}; two contiguous 4-dword runs.
        Frag af;
        const int rowA = waveM * 16 + l15;
        af.q[0] = *(const U4*)&Al[buf][rowA][hi << 2];
        af.q[1] = *(const U4*)&Al[buf][rowA][8 + (hi << 2)];

        // B fragments: same pair layout as A: [col][hi*4 .. +3] and [col][8+hi*4 .. +3].
        // Separate variables -> all 8 loads issue ahead of the WMMA chain.
        const int col0 = waveN * 64 + l15;
        Frag bf0, bf1, bf2, bf3;
        bf0.q[0] = *(const U4*)&Bl[buf][col0     ][hi << 2];
        bf0.q[1] = *(const U4*)&Bl[buf][col0     ][8 + (hi << 2)];
        bf1.q[0] = *(const U4*)&Bl[buf][col0 + 16][hi << 2];
        bf1.q[1] = *(const U4*)&Bl[buf][col0 + 16][8 + (hi << 2)];
        bf2.q[0] = *(const U4*)&Bl[buf][col0 + 32][hi << 2];
        bf2.q[1] = *(const U4*)&Bl[buf][col0 + 32][8 + (hi << 2)];
        bf3.q[0] = *(const U4*)&Bl[buf][col0 + 48][hi << 2];
        bf3.q[1] = *(const U4*)&Bl[buf][col0 + 48][8 + (hi << 2)];

        acc0 = __builtin_amdgcn_wmma_f32_16x16x32_f16(
            false, af.v, false, bf0.v, (short)0, acc0, false, false);
        acc1 = __builtin_amdgcn_wmma_f32_16x16x32_f16(
            false, af.v, false, bf1.v, (short)0, acc1, false, false);
        acc2 = __builtin_amdgcn_wmma_f32_16x16x32_f16(
            false, af.v, false, bf2.v, (short)0, acc2, false, false);
        acc3 = __builtin_amdgcn_wmma_f32_16x16x32_f16(
            false, af.v, false, bf3.v, (short)0, acc3, false, false);
        // No trailing barrier: the wait+barrier at the top of the next
        // iteration separates these LDS reads from the next overwrite.
    }

    // Epilogue: bias + ReLU, store f16.  C/D layout: lane<16 -> M=r, N=lane;
    // lane>=16 -> M=r+8, N=lane-16.
    v8f accs[4] = {acc0, acc1, acc2, acc3};
#pragma unroll
    for (int nt = 0; nt < 4; ++nt) {
        const int col = colBase + waveN * 64 + nt * 16 + l15;
        const float bv = bias[col];
#pragma unroll
        for (int r = 0; r < 8; ++r) {
            const int row = rowBase + waveM * 16 + hi * 8 + r;
            float v = accs[nt][r] + bv;
            v = v > 0.f ? v : 0.f;
            Cout[(size_t)row * Nn + col] = (_Float16)v;
        }
    }
}

// ---------------------------------------------------------------------------
// Head: out[b] = dot(h3[b,:], head_w) + head_b   (f32 reduction)
// ---------------------------------------------------------------------------
__global__ __launch_bounds__(256) void onn_head_kernel(const _Float16* __restrict__ h3,
                                                       const float* __restrict__ hw,
                                                       const float* __restrict__ hb,
                                                       float* __restrict__ out) {
    __shared__ float red[256];
    const int b = blockIdx.x;
    float s = 0.f;
    for (int n = threadIdx.x; n < HDIM; n += 256)
        s += (float)h3[(size_t)b * HDIM + n] * hw[n];
    red[threadIdx.x] = s;
    __syncthreads();
    for (int st = 128; st > 0; st >>= 1) {
        if (threadIdx.x < st) red[threadIdx.x] += red[threadIdx.x + st];
        __syncthreads();
    }
    if (threadIdx.x == 0) out[b] = red[0] + hb[0];
}

// ---------------------------------------------------------------------------
extern "C" void kernel_launch(void* const* d_in, const int* in_sizes, int n_in,
                              void* d_out, int out_size, void* d_ws, size_t ws_size,
                              hipStream_t stream) {
    (void)in_sizes; (void)n_in; (void)out_size; (void)ws_size;
    const int*   x     = (const int*)d_in[0];
    const int*   masks = (const int*)d_in[1];
    const float* W1 = (const float*)d_in[2];
    const float* b1 = (const float*)d_in[3];
    const float* W2 = (const float*)d_in[4];
    const float* b2 = (const float*)d_in[5];
    const float* W3 = (const float*)d_in[6];
    const float* b3 = (const float*)d_in[7];
    const float* hw = (const float*)d_in[8];
    const float* hb = (const float*)d_in[9];
    float* out = (float*)d_out;

    char* ws = (char*)d_ws;
    size_t off = 0;
    auto alloc = [&](size_t bytes) -> void* {
        off = (off + 255) & ~(size_t)255;
        void* p = ws + off;
        off += bytes;
        return p;
    };
    _Float16* feats = (_Float16*)alloc((size_t)BATCH * TFEATS * sizeof(_Float16));
    unsigned* W1pk  = (unsigned*)alloc((size_t)(TFEATS / 2) * HDIM * sizeof(unsigned));
    unsigned* W2pk  = (unsigned*)alloc((size_t)(HDIM / 2)  * HDIM * sizeof(unsigned));
    unsigned* W3pk  = (unsigned*)alloc((size_t)(HDIM / 2)  * HDIM * sizeof(unsigned));
    _Float16* h1    = (_Float16*)alloc((size_t)BATCH * HDIM * sizeof(_Float16));
    _Float16* h2    = (_Float16*)alloc((size_t)BATCH * HDIM * sizeof(_Float16));
    _Float16* h3    = (_Float16*)alloc((size_t)BATCH * HDIM * sizeof(_Float16));

    // Weight f32 -> chunk-swizzled packed f16 (bandwidth-bound, ~10us @23.3TB/s)
    onn_pack_w_kernel<<<dim3(HDIM / 256, TFEATS / 32), 256, 0, stream>>>(W1, W1pk, HDIM);
    onn_pack_w_kernel<<<dim3(HDIM / 256, HDIM / 32),   256, 0, stream>>>(W2, W2pk, HDIM);
    onn_pack_w_kernel<<<dim3(HDIM / 256, HDIM / 32),   256, 0, stream>>>(W3, W3pk, HDIM);

    // Bit-triangle feature stage
    onn_feats_kernel<<<BATCH, 256, 0, stream>>>(x, masks, feats);

    // MLP: three WMMA GEMMs + head
    onn_gemm_kernel<<<dim3(HDIM / 256, BATCH / 32), 256, 0, stream>>>(
        (const unsigned*)feats, W1pk, b1, h1, HDIM, TFEATS);
    onn_gemm_kernel<<<dim3(HDIM / 256, BATCH / 32), 256, 0, stream>>>(
        (const unsigned*)h1, W2pk, b2, h2, HDIM, HDIM);
    onn_gemm_kernel<<<dim3(HDIM / 256, BATCH / 32), 256, 0, stream>>>(
        (const unsigned*)h2, W3pk, b3, h3, HDIM, HDIM);
    onn_head_kernel<<<BATCH, 256, 0, stream>>>(h3, hw, hb, out);
}
